// Quantizer_89369679495715
// MI455X (gfx1250) — matrix operand order repaired and checked by reading, not compiled, not run on hardware
//
#include <hip/hip_runtime.h>
#include <hip/hip_bf16.h>

// Problem sizes (from reference): B=32, T=2048, D=256, V=2048
constexpr int ND   = 65536;   // N = B*T rows
constexpr int ED   = 256;     // embedding dim (K of the GEMM)
constexpr int NV   = 2048;    // number of codes (N of the GEMM)
constexpr int LSTR = 264;     // LDS row stride in bf16 elems (pad 8 -> bank-conflict free)

// Output layout in d_out (floats), concat of reference return tuple:
constexpr int OFF_ZQ   = 1;
constexpr int OFF_IDX  = 1 + ND * ED;        // 16777217
constexpr int OFF_PROB = OFF_IDX + ND;       // 16842753

// Workspace layout (4-byte units)
constexpr int WS_E2   = 0;                   // float[NV]
constexpr int WS_IDX  = WS_E2 + NV;          // int[ND]
constexpr int WS_HIST = WS_IDX + ND;         // int[NV]
constexpr int WS_PART = WS_HIST + NV;        // float[ND/8]
constexpr int NPART   = ND / 8;
constexpr int WS_EHI  = WS_PART + NPART;     // bf16[NV*ED]  (262144 dwords)
constexpr int WS_ELO  = WS_EHI + (NV * ED) / 2; // bf16[NV*ED]

typedef __attribute__((ext_vector_type(16))) __bf16 v16bf;
typedef __attribute__((ext_vector_type(8)))  float  v8f;

union BF16x16 { v16bf v; uint4 u[2]; };

// ---------------- Kernel 1: ||e_v||^2 per code row, and zero the histogram --
__global__ void k_e2_hist0(const float* __restrict__ e,
                           float* __restrict__ e2,
                           int* __restrict__ hist) {
  int gid = blockIdx.x * blockDim.x + threadIdx.x;
  if (gid < NV) hist[gid] = 0;
  int wid = threadIdx.x >> 5, lane = threadIdx.x & 31;
  int row = blockIdx.x * 8 + wid;               // grid 256 blocks -> rows 0..2047
  const float* er = e + (size_t)row * ED + lane * 8;
  float4 a = ((const float4*)er)[0];
  float4 b = ((const float4*)er)[1];
  float s = a.x*a.x + a.y*a.y + a.z*a.z + a.w*a.w
          + b.x*b.x + b.y*b.y + b.z*b.z + b.w*b.w;
  #pragma unroll
  for (int off = 16; off >= 1; off >>= 1) s += __shfl_xor(s, off, 32);
  if (lane == 0) e2[row] = s;
}

// ---------------- Kernel 1b: pre-split e into bf16 hi/lo planes -------------
__global__ void k_cvt_e(const float* __restrict__ e,
                        __bf16* __restrict__ ehi, __bf16* __restrict__ elo) {
  int gid = blockIdx.x * blockDim.x + threadIdx.x;  // 256 blocks x 256 thr
  int base = gid * 8;                               // covers NV*ED = 524288
  float4 f0 = ((const float4*)(e + base))[0];
  float4 f1 = ((const float4*)(e + base))[1];
  float xs[8] = {f0.x,f0.y,f0.z,f0.w, f1.x,f1.y,f1.z,f1.w};
  #pragma unroll
  for (int j = 0; j < 8; ++j) {
    __bf16 h = (__bf16)xs[j];
    ehi[base + j] = h;
    elo[base + j] = (__bf16)(xs[j] - (float)h);
  }
}

// ---------------- Kernel 2: fused GEMM (split-bf16 WMMA) + running argmin ----
// Block = 256 threads = 8 waves; each wave owns 16 rows; block owns 128 rows.
// 16-column code tiles are async-DMA'd (global_load_async_to_lds_b128) into a
// double-buffered LDS stage while WMMAs consume the other buffer.
__global__ __launch_bounds__(256) void k_argmin(
    const float* __restrict__ z,
    const __bf16* __restrict__ ehi, const __bf16* __restrict__ elo,
    const float* __restrict__ e2, int* __restrict__ idx_ws,
    int* __restrict__ hist, float* __restrict__ out_idx_f) {
  __shared__ __bf16 sHi[2][16 * LSTR];
  __shared__ __bf16 sLo[2][16 * LSTR];

  const int t    = threadIdx.x;
  const int wid  = t >> 5;
  const int lane = t & 31;
  const int hsel = (lane >> 4) & 1;   // which K-half this lane carries
  const int n    = lane & 15;         // A: row-in-tile / B,C: column-in-tile
  const int rowbase = blockIdx.x * 128 + wid * 16;

  // ---- Load this wave's 16 rows of z as bf16 hi/lo A-fragments (registers) --
  // 16-bit A 16x32 layout: lanes 0-15: K = c*32 + {0..7, 16..23}
  //                        lanes16-31: K = c*32 + {8..15, 24..31}
  v16bf a_hi[8], a_lo[8];
  {
    const float* zr = z + (size_t)(rowbase + n) * ED;
    #pragma unroll
    for (int c = 0; c < 8; ++c) {
      const int k0 = c * 32 + hsel * 8;
      const int k1 = c * 32 + 16 + hsel * 8;
      float4 f0 = ((const float4*)(zr + k0))[0];
      float4 f1 = ((const float4*)(zr + k0))[1];
      float4 f2 = ((const float4*)(zr + k1))[0];
      float4 f3 = ((const float4*)(zr + k1))[1];
      float xs[16] = {f0.x,f0.y,f0.z,f0.w, f1.x,f1.y,f1.z,f1.w,
                      f2.x,f2.y,f2.z,f2.w, f3.x,f3.y,f3.z,f3.w};
      #pragma unroll
      for (int j = 0; j < 16; ++j) {
        __bf16 h = (__bf16)xs[j];
        a_hi[c][j] = h;
        a_lo[c][j] = (__bf16)(xs[j] - (float)h);
      }
    }
  }

  float minv[8];
  int   mini[8];
  #pragma unroll
  for (int i = 0; i < 8; ++i) { minv[i] = 3.4e38f; mini[i] = 0; }

  // Staging assignment: thread -> (code row r, 16-elem K segment)
  const int stage_r = t >> 4;          // 0..15
  const int stage_k = (t & 15) * 16;   // 0,16,...,240

  // Async DMA one 16-row bf16 tile (hi+lo) into LDS buffer nb.
  auto stage_async = [&](int cb, int nb) {
    const __bf16* gh = ehi + (size_t)(cb + stage_r) * ED + stage_k;
    const __bf16* gl = elo + (size_t)(cb + stage_r) * ED + stage_k;
    unsigned dh = (unsigned)(uintptr_t)&sHi[nb][stage_r * LSTR + stage_k];
    unsigned dl = (unsigned)(uintptr_t)&sLo[nb][stage_r * LSTR + stage_k];
    // INST_OFFSET is added to BOTH the global and the LDS address (ISA 10.x),
    // so offset:16 moves the second 16B chunk of the 32B segment.
    asm volatile(
      "global_load_async_to_lds_b128 %0, %2, off\n\t"
      "global_load_async_to_lds_b128 %0, %2, off offset:16\n\t"
      "global_load_async_to_lds_b128 %1, %3, off\n\t"
      "global_load_async_to_lds_b128 %1, %3, off offset:16"
      :: "v"(dh), "v"(dl), "v"(gh), "v"(gl) : "memory");
  };

  // Prologue: fill buffer 0 with tile 0.
  stage_async(0, 0);
  asm volatile("s_wait_asynccnt 0x0" ::: "memory");
  __syncthreads();

  int it = 0;
  for (int cb = 0; cb < NV; cb += 16, ++it) {
    const int cur = it & 1;
    if (cb + 16 < NV) stage_async(cb + 16, cur ^ 1);  // overlap with WMMAs

    // ---- 16x16 tile: K=256 via 8 chunks x 3 split-bf16 WMMAs, 2 acc chains --
    v8f acc0 = {0.f,0.f,0.f,0.f,0.f,0.f,0.f,0.f};
    v8f acc1 = {0.f,0.f,0.f,0.f,0.f,0.f,0.f,0.f};
    #pragma unroll
    for (int c = 0; c < 8; ++c) {
      const int kb = c * 32 + hsel * 8;   // B: lanes 0-15 K {0..7,16..23}, 16-31 K {8..15,24..31}
      const uint4* ph = (const uint4*)&sHi[cur][n * LSTR + kb];
      const uint4* pl = (const uint4*)&sLo[cur][n * LSTR + kb];
      BF16x16 bh, bl;
      bh.u[0] = ph[0]; bh.u[1] = ph[2];
      bl.u[0] = pl[0]; bl.u[1] = pl[2];
      acc0 = __builtin_amdgcn_wmma_f32_16x16x32_bf16(false, a_hi[c], false, bh.v,
                                                     (short)0, acc0, false, false);
      acc1 = __builtin_amdgcn_wmma_f32_16x16x32_bf16(false, a_hi[c], false, bl.v,
                                                     (short)0, acc1, false, false);
      acc1 = __builtin_amdgcn_wmma_f32_16x16x32_bf16(false, a_lo[c], false, bh.v,
                                                     (short)0, acc1, false, false);
    }

    // dist(m, v) = ||e_v||^2 - 2 z_m . e_v   (||z||^2 dropped: row-constant)
    const float e2v = e2[cb + n];
    const int   vv  = cb + n;            // this lane's column is fixed per tile
    #pragma unroll
    for (int i = 0; i < 8; ++i) {
      float dist = __builtin_fmaf(-2.0f, acc0[i] + acc1[i], e2v);
      if (dist < minv[i]) { minv[i] = dist; mini[i] = vv; }
    }

    // Publish next tile: my async writes done, then block-wide visibility.
    asm volatile("s_wait_asynccnt 0x0" ::: "memory");
    __syncthreads();
  }

  // ---- cross-lane (16-lane group) min+argmin, first-index tie-break ----
  #pragma unroll
  for (int i = 0; i < 8; ++i) {
    float bv = minv[i]; int bi = mini[i];
    #pragma unroll
    for (int off = 8; off >= 1; off >>= 1) {
      float ov = __shfl_xor(bv, off, 32);
      int   oi = __shfl_xor(bi, off, 32);
      if (ov < bv || (ov == bv && oi < bi)) { bv = ov; bi = oi; }
    }
    if (n == 0) {  // lanes 0 and 16 write rows i and i+8 of this wave's tile
      int row = rowbase + i + hsel * 8;
      idx_ws[row]    = bi;
      out_idx_f[row] = (float)bi;
      atomicAdd(&hist[bi], 1);
    }
  }
}

// ---------------- Kernel 3: gather z_q = e[idx], per-block loss partials ----
__global__ void k_gather_loss(const float* __restrict__ z,
                              const float* __restrict__ e,
                              const int* __restrict__ idx_ws,
                              float* __restrict__ zq_out,   // d_out + 1 (4B-aligned only!)
                              float* __restrict__ partials) {
  __shared__ float wsum[8];
  int wid = threadIdx.x >> 5, lane = threadIdx.x & 31;
  int row = blockIdx.x * 8 + wid;
  int ci  = idx_ws[row];
  const float* zr = z + (size_t)row * ED + lane * 8;
  const float* er = e + (size_t)ci  * ED + lane * 8;
  float4 z0 = ((const float4*)zr)[0], z1 = ((const float4*)zr)[1];
  float4 e0 = ((const float4*)er)[0], e1 = ((const float4*)er)[1];
  float* o = zq_out + (size_t)row * ED + lane * 8;
  // zq_out base is d_out+1 -> only 4-byte aligned; scalar b32 stores
  o[0] = e0.x; o[1] = e0.y; o[2] = e0.z; o[3] = e0.w;
  o[4] = e1.x; o[5] = e1.y; o[6] = e1.z; o[7] = e1.w;
  float dx;
  float s = 0.f;
  dx = z0.x - e0.x; s += dx * dx;  dx = z0.y - e0.y; s += dx * dx;
  dx = z0.z - e0.z; s += dx * dx;  dx = z0.w - e0.w; s += dx * dx;
  dx = z1.x - e1.x; s += dx * dx;  dx = z1.y - e1.y; s += dx * dx;
  dx = z1.z - e1.z; s += dx * dx;  dx = z1.w - e1.w; s += dx * dx;
  #pragma unroll
  for (int off = 16; off >= 1; off >>= 1) s += __shfl_xor(s, off, 32);
  if (lane == 0) wsum[wid] = s;
  __syncthreads();
  if (threadIdx.x == 0) {
    float tot = 0.f;
    #pragma unroll
    for (int i = 0; i < 8; ++i) tot += wsum[i];   // fixed order: deterministic
    partials[blockIdx.x] = tot;
  }
}

// ---------------- Kernel 4: deterministic loss reduction + codebook prob ----
__global__ void k_final(const float* __restrict__ partials,
                        const int* __restrict__ hist,
                        float* __restrict__ out) {
  __shared__ float red[256];
  int t = threadIdx.x;
  float s = 0.f;
  for (int j = t; j < NPART; j += 256) s += partials[j];  // fixed order
  red[t] = s;
  __syncthreads();
  for (int st = 128; st >= 1; st >>= 1) {
    if (t < st) red[t] += red[t + st];
    __syncthreads();
  }
  if (t == 0) out[0] = red[0] * (1.0f / 16777216.0f);     // mean over N*D
  // hist.sum() == 65536 exactly; fp32 (65536 + 1e-8) == 65536
  for (int v = t; v < NV; v += 256)
    out[OFF_PROB + v] = (float)hist[v] * (1.0f / 65536.0f);
}

extern "C" void kernel_launch(void* const* d_in, const int* in_sizes, int n_in,
                              void* d_out, int out_size, void* d_ws, size_t ws_size,
                              hipStream_t stream) {
  const float* z = (const float*)d_in[0];          // (B,T,D) fp32
  const float* e = (const float*)d_in[1];          // (V,D)   fp32
  float* out  = (float*)d_out;

  float*  e2   = (float*)d_ws + WS_E2;
  int*    idxw = (int*)d_ws + WS_IDX;
  int*    hist = (int*)d_ws + WS_HIST;
  float*  part = (float*)d_ws + WS_PART;
  __bf16* ehi  = (__bf16*)((int*)d_ws + WS_EHI);
  __bf16* elo  = (__bf16*)((int*)d_ws + WS_ELO);

  k_e2_hist0   <<<NV / 8,    256, 0, stream>>>(e, e2, hist);
  k_cvt_e      <<<256,       256, 0, stream>>>(e, ehi, elo);
  k_argmin     <<<ND / 128,  256, 0, stream>>>(z, ehi, elo, e2, idxw, hist, out + OFF_IDX);
  k_gather_loss<<<ND / 8,    256, 0, stream>>>(z, e, idxw, out + OFF_ZQ, part);
  k_final      <<<1,         256, 0, stream>>>(part, hist, out);
}
